// MultiEdgeClassifier_83614423318732
// MI455X (gfx1250) — matrix operand order repaired and checked by reading, not compiled, not run on hardware
//
#include <hip/hip_runtime.h>
#include <hip/hip_bf16.h>

#define N_NODES 100000
#define N_EDGES 1600000
#define N_EOUT  400000
#define HID     64
#define NLAYERS 6
#define BN_EPS  1e-5f

typedef __attribute__((ext_vector_type(16))) _Float16 v16h;
typedef __attribute__((ext_vector_type(2)))  _Float16 v2h;
typedef __attribute__((ext_vector_type(8)))  float    v8f;

// ---------------- degree / normalization ----------------
__global__ void init_deg_kernel(float* __restrict__ deg, int n) {
  int i = blockIdx.x * blockDim.x + threadIdx.x;
  if (i < n) deg[i] = 1.0f;  // self-loop contributes 1 to every node's degree
}

__global__ void accum_deg_kernel(const int* __restrict__ dst, float* __restrict__ deg, int nE) {
  int e = blockIdx.x * blockDim.x + threadIdx.x;
  if (e < nE) unsafeAtomicAdd(&deg[dst[e]], 1.0f);
}

__global__ void finalize_dinv_kernel(float* __restrict__ deg, int n) {
  int i = blockIdx.x * blockDim.x + threadIdx.x;
  if (i < n) deg[i] = rsqrtf(deg[i]);  // deg >= 1 always (self loops)
}

__global__ void compute_norm_kernel(const int* __restrict__ src, const int* __restrict__ dst,
                                    const float* __restrict__ dinv, float* __restrict__ norm, int nE) {
  int e = blockIdx.x * blockDim.x + threadIdx.x;
  if (e < nE) norm[e] = dinv[src[e]] * dinv[dst[e]];
}

// ---------------- node embedding: x_embed = x @ W_emb + b_emb ----------------
__global__ void embed_kernel(const float* __restrict__ x, const float* __restrict__ Wemb,
                             const float* __restrict__ bemb, float* __restrict__ xe, int nRows) {
  __shared__ float sW[16 * 64];
  __shared__ float sb[64];
  for (int i = threadIdx.x; i < 16 * 64; i += blockDim.x) sW[i] = Wemb[i];
  if (threadIdx.x < 64) sb[threadIdx.x] = bemb[threadIdx.x];
  __syncthreads();
  int c = threadIdx.x & 63;
  int r = blockIdx.x * (blockDim.x >> 6) + (threadIdx.x >> 6);
  if (r >= nRows) return;
  float acc = sb[c];
#pragma unroll
  for (int k = 0; k < 16; ++k) acc += x[(size_t)r * 16 + k] * sW[k * 64 + c];
  xe[(size_t)r * 64 + c] = acc;
}

// ---------------- per-layer weight conversion (f32 -> f16) + BN-stat zeroing ----------------
__global__ void convw_kernel(const float* __restrict__ W, _Float16* __restrict__ Wh,
                             float* __restrict__ stats) {
  int i = blockIdx.x * blockDim.x + threadIdx.x;
  if (i < HID * HID) Wh[i] = (_Float16)W[i];
  if (i < 2 * HID) stats[i] = 0.0f;
}

// ---------------- WMMA GEMM: hw = x_embed(Nx64,f32->f16) @ W(64x64,f16), f32 acc ----------------
__global__ void gemm_xw_kernel(const float* __restrict__ xe, const _Float16* __restrict__ Wh,
                               float* __restrict__ hw, int nRowBlocks) {
  int wave = (blockIdx.x * blockDim.x + threadIdx.x) >> 5;  // wave32
  if (wave >= nRowBlocks) return;                           // wave-uniform guard (EXEC all-1)
  int lane = threadIdx.x & 31;
  int g = lane >> 4;   // half-wave select
  int m = lane & 15;   // row within tile (A), col within tile (C/D)
  int row = wave * 16 + m;

  // A fragments: 16x32 f16, two K-chunks (K=0..31, 32..63), converted inline from f32
  v16h a[2];
#pragma unroll
  for (int s = 0; s < 2; ++s) {
#pragma unroll
    for (int v = 0; v < 8; ++v) {
      int k0 = 32 * s + ((v < 4) ? (8 * g + 2 * v) : (16 + 8 * g + 2 * (v - 4)));
      float2 f = *reinterpret_cast<const float2*>(xe + (size_t)row * 64 + k0);
      a[s][2 * v]     = (_Float16)f.x;
      a[s][2 * v + 1] = (_Float16)f.y;
    }
  }

  int kb = lane & 15;  // B column within 16-wide N tile
  v8f acc[4] = {};
#pragma unroll
  for (int s = 0; s < 2; ++s) {
    int krow = 32 * s;
#pragma unroll
    for (int nt = 0; nt < 4; ++nt) {   // four 16-wide N tiles -> 64 output cols
      v16h b;
#pragma unroll
      for (int v = 0; v < 8; ++v) {
        int k0 = krow + ((v < 4) ? (8 * g + 2 * v) : (16 + 8 * g + 2 * (v - 4)));
        b[2 * v]     = Wh[k0 * 64 + nt * 16 + kb];
        b[2 * v + 1] = Wh[(k0 + 1) * 64 + nt * 16 + kb];
      }
      acc[nt] = __builtin_amdgcn_wmma_f32_16x16x32_f16(
          /*neg_a=*/false, a[s], /*neg_b=*/false, b,
          /*c_mod=*/(short)0, acc[nt], /*reuse_a=*/false, /*reuse_b=*/false);
    }
  }

  // D layout: element v is D[8g+v][lane&15]
#pragma unroll
  for (int nt = 0; nt < 4; ++nt)
#pragma unroll
    for (int v = 0; v < 8; ++v)
      hw[(size_t)(wave * 16 + 8 * g + v) * 64 + nt * 16 + m] = acc[nt][v];
}

// ---------------- agg init + fused self-loop: agg[i] = hw[i] * dinv[row]^2 ----------------
__global__ void selfloop_init_kernel(const float* __restrict__ hw, const float* __restrict__ dinv,
                                     float* __restrict__ agg, int total4) {
  int i = blockIdx.x * blockDim.x + threadIdx.x;  // over float4 units (16 per row)
  if (i >= total4) return;
  int rowi = i >> 4;
  float dv = dinv[rowi];
  float d2 = dv * dv;
  float4 v = reinterpret_cast<const float4*>(hw)[i];
  v.x *= d2; v.y *= d2; v.z *= d2; v.w *= d2;
  reinterpret_cast<float4*>(agg)[i] = v;
}

// ---------------- edge scatter: agg[dst] += norm * hw[src], 16 lanes/edge, f32 HW atomics ----
__global__ void scatter_kernel(const float* __restrict__ hw, const int* __restrict__ src,
                               const int* __restrict__ dst, const float* __restrict__ norm,
                               float* __restrict__ agg, int nE) {
  long long tid = (long long)blockIdx.x * blockDim.x + threadIdx.x;
  int e = (int)(tid >> 4);
  if (e >= nE) return;
  int sub = (int)(tid & 15);
  int s = src[e], d = dst[e];
  float nrm = norm[e];
  float4 hv = *reinterpret_cast<const float4*>(hw + (size_t)s * 64 + sub * 4);
  float* out = agg + (size_t)d * 64 + sub * 4;
  unsafeAtomicAdd(out + 0, hv.x * nrm);
  unsafeAtomicAdd(out + 1, hv.y * nrm);
  unsafeAtomicAdd(out + 2, hv.z * nrm);
  unsafeAtomicAdd(out + 3, hv.w * nrm);
}

// ---------------- BN stats: sum / sumsq of (agg + b) per channel ----------------
__global__ void bn_stats_kernel(const float* __restrict__ agg, const float* __restrict__ b,
                                float* __restrict__ stats, int nRows) {
  __shared__ float s_sum[256];
  __shared__ float s_sq[256];
  int c = threadIdx.x & 63;
  int rstart = blockIdx.x * (blockDim.x >> 6) + (threadIdx.x >> 6);
  int rstride = gridDim.x * (blockDim.x >> 6);
  float bc = b[c];
  float sum = 0.0f, sq = 0.0f;
  for (int r = rstart; r < nRows; r += rstride) {
    float v = agg[(size_t)r * 64 + c] + bc;
    sum += v;
    sq += v * v;
  }
  s_sum[threadIdx.x] = sum;
  s_sq[threadIdx.x] = sq;
  __syncthreads();
  if (threadIdx.x < 128) {
    s_sum[threadIdx.x] += s_sum[threadIdx.x + 128];
    s_sq[threadIdx.x]  += s_sq[threadIdx.x + 128];
  }
  __syncthreads();
  if (threadIdx.x < 64) {
    unsafeAtomicAdd(&stats[threadIdx.x],      s_sum[threadIdx.x] + s_sum[threadIdx.x + 64]);
    unsafeAtomicAdd(&stats[64 + threadIdx.x], s_sq[threadIdx.x]  + s_sq[threadIdx.x + 64]);
  }
}

// ---------------- fused BN + ReLU + residual: xe += relu(bn(agg + b)) ----------------
__global__ void bn_update_kernel(const float* __restrict__ agg, const float* __restrict__ stats,
                                 const float* __restrict__ b, const float* __restrict__ gamma,
                                 const float* __restrict__ beta, float* __restrict__ xe, int total) {
  __shared__ float s_scale[64];
  __shared__ float s_shift[64];
  if (threadIdx.x < 64) {
    const float invN = 1.0f / (float)N_NODES;
    float mean = stats[threadIdx.x] * invN;
    float var = stats[64 + threadIdx.x] * invN - mean * mean;
    var = var > 0.0f ? var : 0.0f;
    float sc = gamma[threadIdx.x] * rsqrtf(var + BN_EPS);
    s_scale[threadIdx.x] = sc;
    s_shift[threadIdx.x] = (b[threadIdx.x] - mean) * sc + beta[threadIdx.x];
  }
  __syncthreads();
  for (int i = blockIdx.x * blockDim.x + threadIdx.x; i < total; i += gridDim.x * blockDim.x) {
    int c = i & 63;
    float h = agg[i] * s_scale[c] + s_shift[c];
    h = h > 0.0f ? h : 0.0f;
    xe[i] += h;
  }
}

// ---------------- edge head: out = [xe[src], xe[dst]] @ fc_W + fc_b (one wave/edge) ----------
__global__ void edge_fc_kernel(const float* __restrict__ xe, const int* __restrict__ esrc,
                               const int* __restrict__ edst, const float* __restrict__ fcW,
                               const float* __restrict__ fcb, float* __restrict__ out, int nE) {
  int wave = (blockIdx.x * blockDim.x + threadIdx.x) >> 5;
  if (wave >= nE) return;
  int lane = threadIdx.x & 31;
  int s = esrc[wave], d = edst[wave];
  float xs0 = xe[(size_t)s * 64 + lane];
  float xs1 = xe[(size_t)s * 64 + 32 + lane];
  float xd0 = xe[(size_t)d * 64 + lane];
  float xd1 = xe[(size_t)d * 64 + 32 + lane];
  float p0 = xs0 * fcW[lane * 2]     + xs1 * fcW[(lane + 32) * 2] +
             xd0 * fcW[(lane + 64) * 2] + xd1 * fcW[(lane + 96) * 2];
  float p1 = xs0 * fcW[lane * 2 + 1] + xs1 * fcW[(lane + 32) * 2 + 1] +
             xd0 * fcW[(lane + 64) * 2 + 1] + xd1 * fcW[(lane + 96) * 2 + 1];
#pragma unroll
  for (int o = 16; o > 0; o >>= 1) {
    p0 += __shfl_down(p0, o, 32);
    p1 += __shfl_down(p1, o, 32);
  }
  if (lane == 0) {
    out[(size_t)wave * 2]     = p0 + fcb[0];
    out[(size_t)wave * 2 + 1] = p1 + fcb[1];
  }
}

extern "C" void kernel_launch(void* const* d_in, const int* in_sizes, int n_in,
                              void* d_out, int out_size, void* d_ws, size_t ws_size,
                              hipStream_t stream) {
  (void)in_sizes; (void)n_in; (void)out_size; (void)ws_size;

  const float* x        = (const float*)d_in[0];
  const int*   ei       = (const int*)d_in[1];   // [2, E]
  const int*   eo       = (const int*)d_in[2];   // [2, E_OUT]
  const float* W_emb    = (const float*)d_in[3];
  const float* b_emb    = (const float*)d_in[4];
  const float* conv_W   = (const float*)d_in[5]; // [L, 64, 64]
  const float* conv_b   = (const float*)d_in[6]; // [L, 64]
  const float* bn_gamma = (const float*)d_in[7];
  const float* bn_beta  = (const float*)d_in[8];
  const float* fc_W     = (const float*)d_in[9]; // [128, 2]
  const float* fc_b     = (const float*)d_in[10];
  float* out = (float*)d_out;

  const int* src = ei;
  const int* dst = ei + N_EDGES;
  const int* esrc = eo;
  const int* edst = eo + N_EOUT;

  // carve workspace
  size_t off = 0;
  auto take = [&](size_t bytes) -> void* {
    void* p = (char*)d_ws + off;
    off += (bytes + 511) & ~(size_t)511;
    return p;
  };
  float*    dinv  = (float*)take((size_t)N_NODES * 4);
  float*    norm  = (float*)take((size_t)N_EDGES * 4);
  float*    xe    = (float*)take((size_t)N_NODES * HID * 4);
  float*    hw    = (float*)take((size_t)N_NODES * HID * 4);
  float*    agg   = (float*)take((size_t)N_NODES * HID * 4);
  _Float16* Wh    = (_Float16*)take((size_t)HID * HID * 2);
  float*    stats = (float*)take((size_t)2 * HID * 4);

  const int TPB = 256;

  // degree + symmetric normalization (shared across layers)
  init_deg_kernel<<<(N_NODES + TPB - 1) / TPB, TPB, 0, stream>>>(dinv, N_NODES);
  accum_deg_kernel<<<(N_EDGES + TPB - 1) / TPB, TPB, 0, stream>>>(dst, dinv, N_EDGES);
  finalize_dinv_kernel<<<(N_NODES + TPB - 1) / TPB, TPB, 0, stream>>>(dinv, N_NODES);
  compute_norm_kernel<<<(N_EDGES + TPB - 1) / TPB, TPB, 0, stream>>>(src, dst, dinv, norm, N_EDGES);

  // embedding
  embed_kernel<<<(N_NODES + 3) / 4, TPB, 0, stream>>>(x, W_emb, b_emb, xe, N_NODES);

  const int rowBlocks = N_NODES / 16;              // 6250 (exact)
  const int gemmBlocks = (rowBlocks + 7) / 8;      // 8 waves/block
  const int total = N_NODES * HID;
  const int total4 = total / 4;

  for (int l = 0; l < NLAYERS; ++l) {
    const float* Wl = conv_W + (size_t)l * HID * HID;
    const float* bl = conv_b + (size_t)l * HID;
    const float* gl = bn_gamma + (size_t)l * HID;
    const float* btl = bn_beta + (size_t)l * HID;

    convw_kernel<<<(HID * HID + TPB - 1) / TPB, TPB, 0, stream>>>(Wl, Wh, stats);
    gemm_xw_kernel<<<gemmBlocks, TPB, 0, stream>>>(xe, Wh, hw, rowBlocks);
    selfloop_init_kernel<<<(total4 + TPB - 1) / TPB, TPB, 0, stream>>>(hw, dinv, agg, total4);
    scatter_kernel<<<(int)(((long long)N_EDGES * 16 + TPB - 1) / TPB), TPB, 0, stream>>>(
        hw, src, dst, norm, agg, N_EDGES);
    bn_stats_kernel<<<400, TPB, 0, stream>>>(agg, bl, stats, N_NODES);
    bn_update_kernel<<<4096, TPB, 0, stream>>>(agg, stats, bl, gl, btl, xe, total);
  }

  edge_fc_kernel<<<(N_EOUT * 32 + TPB - 1) / TPB, TPB, 0, stream>>>(
      xe, esrc, edst, fc_W, fc_b, out, N_EOUT);
}
